// NLayerMambaModel_31533649887307
// MI455X (gfx1250) — compile-verified, matrix operand order checked by reading, not run on hardware
//
#include <hip/hip_runtime.h>
#include <hip/hip_bf16.h>
#include <math.h>

// ---------------- model constants ----------------
#define N_LAYERS 2
#define D_MODEL  1024
#define D_INNER  2048
#define D_STATE  16
#define D_CONV   4
#define DT_RANK  64
#define BATCH    2
#define SEQ      1024
#define MROWS    (BATCH * SEQ)           // 2048 flattened (b,l) rows
#define XDBL_W   (DT_RANK + 2 * D_STATE) // 96

typedef __attribute__((ext_vector_type(16))) __bf16 v16bf;
typedef __attribute__((ext_vector_type(8)))  float  v8f;
typedef __attribute__((ext_vector_type(4)))  float  f4v;

// Load a 16x32 bf16 WMMA fragment at pointer r (row base + kh already folded).
// elements 0-7 = 16 contiguous bytes at r, elements 8-15 = 16 bytes at r+32B.
// Cast to address_space(1) so the backend emits global_load_b128 (not flat).
__device__ __forceinline__ v16bf load_frag_ptr(const __bf16* __restrict__ r) {
  typedef __attribute__((address_space(1))) const f4v gf4;
  union { f4v q[2]; v16bf f; } u;
  const gf4* g = (const gf4*)r;
  u.q[0] = g[0];
  u.q[1] = g[2];
  return u.f;
}

// ---------------- WMMA GEMM:  C(MxN) = A(MxK) * Bw(NxK)^T ----------------
// Wave tile = (WM*16) x (NT*16); 4 waves per block stacked along M.
// K loop unrolled x2 with ping-pong fragment buffers: stage s loads while
// stage 1-s computes -> no register copies, no WMMA->VALU WAR hazards.
// Requires K % 64 == 0 (holds: K in {64, 1024, 2048}).
// EPI: 0 = f32 store, 1 = softplus(val+bias[n]) f32, 2 = val+resid f32,
//      3 = bf16 store only, 4 = f32 store + bf16 store
template <int WM, int NT, int EPI>
__global__ void __launch_bounds__(128, 1)
wmma_gemm_kernel(const __bf16* __restrict__ A, int lda,
                 const __bf16* __restrict__ Bw, int ldb,
                 float* __restrict__ Cf, __bf16* __restrict__ Cbf,
                 int ldc, int K,
                 const float* __restrict__ bias,
                 const float* __restrict__ resid) {
  const int lane = threadIdx.x & 31;
  const int wave = threadIdx.x >> 5;
  const int half = lane >> 4;
  const int l16  = lane & 15;
  const int m0   = (blockIdx.y * (blockDim.x >> 5) + wave) * (WM * 16);
  const int n0   = blockIdx.x * (NT * 16);
  const int kh0  = half * 8;

  v8f acc[WM][NT];
#pragma unroll
  for (int w = 0; w < WM; ++w)
#pragma unroll
    for (int t = 0; t < NT; ++t) acc[w][t] = v8f{};

  // per-fragment row pointers (advanced by 32 elements per stage-load)
  const __bf16* pa[WM];
  const __bf16* pb[NT];
#pragma unroll
  for (int w = 0; w < WM; ++w)
    pa[w] = A + (size_t)(m0 + w * 16 + l16) * lda + kh0;
#pragma unroll
  for (int t = 0; t < NT; ++t)
    pb[t] = Bw + (size_t)(n0 + t * 16 + l16) * ldb + kh0;

  v16bf afr[2][WM], bfr[2][NT];

  auto load_stage = [&](int s) {
#pragma unroll
    for (int w = 0; w < WM; ++w) { afr[s][w] = load_frag_ptr(pa[w]); pa[w] += 32; }
#pragma unroll
    for (int t = 0; t < NT; ++t) { bfr[s][t] = load_frag_ptr(pb[t]); pb[t] += 32; }
  };
  auto mma_stage = [&](int s) {
#pragma unroll
    for (int w = 0; w < WM; ++w)
#pragma unroll
      for (int t = 0; t < NT; ++t)
        acc[w][t] = __builtin_amdgcn_wmma_f32_16x16x32_bf16(
            false, afr[s][w], false, bfr[s][t], (short)0, acc[w][t], false, false);
  };

  load_stage(0);
  for (int k = 0; k < K; k += 64) {
    load_stage(1);          // k+32 fragments in flight ...
    mma_stage(0);           // ... while stage-0 WMMAs execute
    if (k + 64 < K) load_stage(0);
    mma_stage(1);
  }

  // epilogue
#pragma unroll
  for (int w = 0; w < WM; ++w)
#pragma unroll
    for (int t = 0; t < NT; ++t) {
      const int n = n0 + t * 16 + l16;
#pragma unroll
      for (int v = 0; v < 8; ++v) {
        const int m = m0 + w * 16 + half * 8 + v;
        float val = acc[w][t][v];
        if constexpr (EPI == 1) {
          val += bias[n];
          val = (val > 20.f) ? val : log1pf(__expf(val));
        }
        if constexpr (EPI == 2) val += resid[(size_t)m * ldc + n];
        if constexpr (EPI == 0 || EPI == 1 || EPI == 2 || EPI == 4)
          Cf[(size_t)m * ldc + n] = val;
        if constexpr (EPI == 3 || EPI == 4)
          Cbf[(size_t)m * ldc + n] = (__bf16)val;
      }
    }
}

// ------------- fp32 -> bf16 bulk conversion -------------
__global__ void cvt_bf16_kernel(const float* __restrict__ in,
                                __bf16* __restrict__ out, int n) {
  const int i = blockIdx.x * blockDim.x + threadIdx.x;
  if (i < n) out[i] = (__bf16)in[i];
}

// ------------- causal depthwise conv1d (k=4) + SiLU -> u (f32 + bf16) ------
__global__ void conv_silu_kernel(const float* __restrict__ xz,
                                 const float* __restrict__ w,
                                 const float* __restrict__ b,
                                 float* __restrict__ uf,
                                 __bf16* __restrict__ ubf) {
  const int idx = blockIdx.x * blockDim.x + threadIdx.x;
  if (idx >= MROWS * D_INNER) return;
  const int d  = idx % D_INNER;
  const int l  = (idx / D_INNER) % SEQ;
  const int bb = idx / (D_INNER * SEQ);
  const float* wr = w + d * D_CONV;
  float s = b[d];
#pragma unroll
  for (int j = 0; j < D_CONV; ++j) {
    const int lp = l - (D_CONV - 1) + j;
    if (lp >= 0)
      s += wr[j] * xz[((size_t)(bb * SEQ + lp)) * (2 * D_INNER) + d];
  }
  s = s / (1.f + __expf(-s));  // silu
  uf[idx]  = s;
  ubf[idx] = (__bf16)s;
}

// ------------- selective scan, fused +u*D and *silu(z); emits bf16 y -------
__global__ void selective_scan_kernel(const float* __restrict__ delta,
                                      const float* __restrict__ u,
                                      const float* __restrict__ xdbl,
                                      const float* __restrict__ xz,
                                      const float* __restrict__ A_log,
                                      const float* __restrict__ Dp,
                                      __bf16* __restrict__ y) {
  const int idx = blockIdx.x * blockDim.x + threadIdx.x;
  if (idx >= BATCH * D_INNER) return;
  const int d = idx % D_INNER;
  const int b = idx / D_INNER;

  float A[D_STATE], h[D_STATE];
#pragma unroll
  for (int n = 0; n < D_STATE; ++n) {
    A[n] = -__expf(A_log[d * D_STATE + n]);
    h[n] = 0.f;
  }
  const float Dv = Dp[d];

  for (int l = 0; l < SEQ; ++l) {
    const size_t row = (size_t)(b * SEQ + l);
    const float dt = delta[row * D_INNER + d];
    const float uu = u[row * D_INNER + d];
    const float du = dt * uu;
    const float* bc = xdbl + row * XDBL_W;
    float acc = 0.f;
#pragma unroll
    for (int n = 0; n < D_STATE; ++n) {
      h[n] = __expf(dt * A[n]) * h[n] + du * bc[DT_RANK + n];
      acc += h[n] * bc[DT_RANK + D_STATE + n];
    }
    const float zz = xz[row * (2 * D_INNER) + D_INNER + d];
    float val = acc + uu * Dv;
    val *= zz / (1.f + __expf(-zz));  // * silu(z)
    y[row * D_INNER + d] = (__bf16)val;
  }
}

// ---------------- host-side orchestration ----------------
extern "C" void kernel_launch(void* const* d_in, const int* in_sizes, int n_in,
                              void* d_out, int out_size, void* d_ws,
                              size_t ws_size, hipStream_t stream) {
  (void)in_sizes; (void)n_in; (void)out_size; (void)ws_size;
  const float* x         = (const float*)d_in[0];
  const float* in_proj_w = (const float*)d_in[1];
  const float* conv_w    = (const float*)d_in[2];
  const float* conv_b    = (const float*)d_in[3];
  const float* x_proj_w  = (const float*)d_in[4];
  const float* dt_proj_w = (const float*)d_in[5];
  const float* dt_proj_b = (const float*)d_in[6];
  const float* A_log     = (const float*)d_in[7];
  const float* Dw        = (const float*)d_in[8];
  const float* out_proj_w= (const float*)d_in[9];
  float* out = (float*)d_out;

  // workspace layout; all activations stay L2-resident (~116 MB << 192 MB)
  char* wsb = (char*)d_ws;
  size_t off = 0;
  auto alloc = [&](size_t bytes) -> void* {
    void* p = wsb + off;
    off += (bytes + 255) & ~(size_t)255;
    return p;
  };
  float*  xz     = (float*) alloc((size_t)MROWS * 2 * D_INNER * 4);
  float*  uf     = (float*) alloc((size_t)MROWS * D_INNER * 4);
  float*  xdblf  = (float*) alloc((size_t)MROWS * XDBL_W * 4);
  float*  delta  = (float*) alloc((size_t)MROWS * D_INNER * 4);
  __bf16* ubf    = (__bf16*)alloc((size_t)MROWS * D_INNER * 2);
  __bf16* xdblbf = (__bf16*)alloc((size_t)MROWS * XDBL_W * 2);
  __bf16* ybf    = (__bf16*)alloc((size_t)MROWS * D_INNER * 2);
  __bf16* xbf    = (__bf16*)alloc((size_t)MROWS * D_MODEL * 2);
  __bf16* xcurbf = (__bf16*)alloc((size_t)MROWS * D_MODEL * 2);
  __bf16* win_bf = (__bf16*)alloc((size_t)N_LAYERS * 2 * D_INNER * D_MODEL * 2);
  __bf16* wx_bf  = (__bf16*)alloc((size_t)N_LAYERS * XDBL_W * D_INNER * 2);
  __bf16* wdt_bf = (__bf16*)alloc((size_t)N_LAYERS * D_INNER * DT_RANK * 2);
  __bf16* wout_bf= (__bf16*)alloc((size_t)N_LAYERS * D_MODEL * D_INNER * 2);

  auto cvt = [&](const float* src, __bf16* dst, int n) {
    cvt_bf16_kernel<<<(n + 255) / 256, 256, 0, stream>>>(src, dst, n);
  };
  cvt(x,          xbf,     MROWS * D_MODEL);
  cvt(in_proj_w,  win_bf,  N_LAYERS * 2 * D_INNER * D_MODEL);
  cvt(x_proj_w,   wx_bf,   N_LAYERS * XDBL_W * D_INNER);
  cvt(dt_proj_w,  wdt_bf,  N_LAYERS * D_INNER * DT_RANK);
  cvt(out_proj_w, wout_bf, N_LAYERS * D_MODEL * D_INNER);

  const dim3 blk(128);  // 4 waves/block; wave tile = 32M x (NT*16)N

  for (int layer = 0; layer < N_LAYERS; ++layer) {
    const __bf16* Xin = (layer == 0) ? xbf : xcurbf;

    // 1) xz = Xin (2048x1024) * W_in^T  -> f32
    wmma_gemm_kernel<2, 4, 0><<<dim3((2 * D_INNER) / 64, MROWS / 128), blk, 0, stream>>>(
        Xin, D_MODEL, win_bf + (size_t)layer * 2 * D_INNER * D_MODEL, D_MODEL,
        xz, nullptr, 2 * D_INNER, D_MODEL, nullptr, nullptr);

    // 2) u = silu(conv(xc)+b) -> f32 (scan) + bf16 (GEMMs)
    conv_silu_kernel<<<(MROWS * D_INNER + 255) / 256, 256, 0, stream>>>(
        xz, conv_w + (size_t)layer * D_INNER * D_CONV,
        conv_b + (size_t)layer * D_INNER, uf, ubf);

    // 3) x_dbl = u * W_x^T (N=96) -> f32 (scan B,C) + bf16 (dt for GEMM3)
    wmma_gemm_kernel<2, 3, 4><<<dim3(XDBL_W / 48, MROWS / 128), blk, 0, stream>>>(
        ubf, D_INNER, wx_bf + (size_t)layer * XDBL_W * D_INNER, D_INNER,
        xdblf, xdblbf, XDBL_W, D_INNER, nullptr, nullptr);

    // 4) delta = softplus(dt (lda=96, K=64) * W_dt^T + b_dt) -> f32
    wmma_gemm_kernel<2, 4, 1><<<dim3(D_INNER / 64, MROWS / 128), blk, 0, stream>>>(
        xdblbf, XDBL_W, wdt_bf + (size_t)layer * D_INNER * DT_RANK, DT_RANK,
        delta, nullptr, D_INNER, DT_RANK,
        dt_proj_b + (size_t)layer * D_INNER, nullptr);

    // 5) selective scan (sequential in L) -> y bf16
    selective_scan_kernel<<<(BATCH * D_INNER) / 64, 64, 0, stream>>>(
        delta, uf, xdblf, xz, A_log + (size_t)layer * D_INNER * D_STATE,
        Dw + (size_t)layer * D_INNER, ybf);

    // 6) out = y * W_out^T; layer0 -> bf16 xcur, layer1 -> f32 out + residual
    if (layer == N_LAYERS - 1) {
      wmma_gemm_kernel<2, 4, 2><<<dim3(D_MODEL / 64, MROWS / 128), blk, 0, stream>>>(
          ybf, D_INNER, wout_bf + (size_t)layer * D_MODEL * D_INNER, D_INNER,
          out, nullptr, D_MODEL, D_INNER, nullptr, x);
    } else {
      wmma_gemm_kernel<2, 4, 3><<<dim3(D_MODEL / 64, MROWS / 128), blk, 0, stream>>>(
          ybf, D_INNER, wout_bf + (size_t)layer * D_MODEL * D_INNER, D_INNER,
          nullptr, xcurbf, D_MODEL, D_INNER, nullptr, nullptr);
    }
  }
}
